// GraphMatchDist_31190052504135
// MI455X (gfx1250) — matrix operand order repaired and checked by reading, not compiled, not run on hardware
//
#include <hip/hip_runtime.h>
#include <hip/hip_bf16.h>

// Problem constants (from reference)
#define NPTS 32768
#define NQ   8192
#define KNN  16
#define LROW 8
#define HD   128
#define CD   20
#define FD   (HD + CD)   // 148

typedef __attribute__((ext_vector_type(16))) _Float16 v16h;
typedef __attribute__((ext_vector_type(8)))  float    v8f;

#define FINF __builtin_inff()

// ---------------------------------------------------------------------------
// WMMA fragment helpers (CDNA5 wave32 layouts, CDNA5 ISA §7.12.2)
// ---------------------------------------------------------------------------

// A-matrix 16x32 f16: lane -> row M = lane&15; K halves split by lane<16.
__device__ inline v16h load_A16x32(const _Float16* p, int ld) {
  int lane = threadIdx.x & 31;
  int m = lane & 15;
  int kb = (lane < 16) ? 0 : 16;
  v16h r;
#pragma unroll
  for (int e = 0; e < 16; ++e) {
    int k = kb + e;
    r[e] = p[m * ld + k];
  }
  return r;
}

// B-matrix 32x16 f16 from K-major source p[k*ld + n]
__device__ inline v16h load_B32x16_KN(const _Float16* p, int ld) {
  int lane = threadIdx.x & 31;
  int n = lane & 15;
  int kb = (lane < 16) ? 0 : 16;
  v16h r;
#pragma unroll
  for (int e = 0; e < 16; ++e) r[e] = p[(kb + e) * ld + n];
  return r;
}

// B-matrix 32x16 f16 from N-major (transposed) source p[n*ld + k]
__device__ inline v16h load_B32x16_NK(const _Float16* p, int ld) {
  int lane = threadIdx.x & 31;
  int n = lane & 15;
  int kb = (lane < 16) ? 0 : 16;
  v16h r;
#pragma unroll
  for (int e = 0; e < 16; ++e) r[e] = p[n * ld + kb + e];
  return r;
}

__device__ inline v8f wmma_f16(v16h a, v16h b, v8f c) {
  return __builtin_amdgcn_wmma_f32_16x16x32_f16(false, a, false, b,
                                                (short)0, c, false, false);
}

// ---------------------------------------------------------------------------
// Kernel 0: convert weights to zero-padded f16 (K dims padded to 32)
// ---------------------------------------------------------------------------
__global__ void prep_kernel(const float* __restrict__ W1,
                            const float* __restrict__ W2,
                            const float* __restrict__ Wl,
                            _Float16* __restrict__ W1h,   // [32][64]
                            _Float16* __restrict__ W2h,   // [64][128]
                            _Float16* __restrict__ Wlh) { // [32][128]
  int t = blockIdx.x * blockDim.x + threadIdx.x;
  int stride = gridDim.x * blockDim.x;
  for (int e = t; e < 32 * 64; e += stride) {
    int r = e >> 6;
    W1h[e] = (_Float16)(r < 10 ? W1[r * 64 + (e & 63)] : 0.f);
  }
  for (int e = t; e < 64 * 128; e += stride) W2h[e] = (_Float16)W2[e];
  for (int e = t; e < 32 * 128; e += stride) {
    int r = e >> 7;
    Wlh[e] = (_Float16)(r < CD ? Wl[r * 128 + (e & 127)] : 0.f);
  }
}

// ---------------------------------------------------------------------------
// Kernel 1: batched KNN.  One wave per query, 8 queries per 256-thread block.
// LDS tile of 1024 support points shared by all 8 waves (cuts L2 traffic 8x).
// Per-lane sorted top-16 in registers; 16-round wave-argmin merge with index
// tiebreak to match jax.lax.top_k semantics.
// ---------------------------------------------------------------------------
#define TILE 1024
__global__ __launch_bounds__(256) void knn_kernel(
    const float* __restrict__ support_xyz, const int* __restrict__ batch_index,
    const int* __restrict__ filtered_index, int* __restrict__ col) {
  __shared__ float sx[TILE], sy[TILE], sz[TILE];
  __shared__ int sb[TILE];
  __shared__ float md[8][32 * 16];
  __shared__ int mi[8][32 * 16];

  int t = threadIdx.x, wave = t >> 5, lane = t & 31;
  int q = blockIdx.x * 8 + wave;
  int fi = filtered_index[q];
  float qx = support_xyz[3 * fi], qy = support_xyz[3 * fi + 1],
        qz = support_xyz[3 * fi + 2];
  int qb = batch_index[fi];
  float qq = qx * qx + qy * qy + qz * qz;

  float bd[16];
  int bi[16];
#pragma unroll
  for (int s = 0; s < 16; ++s) { bd[s] = FINF; bi[s] = 0x7fffffff; }

  for (int base = 0; base < NPTS; base += TILE) {
    __syncthreads();
    for (int e = t; e < TILE; e += 256) {
      int j = base + e;
      sx[e] = support_xyz[3 * j];
      sy[e] = support_xyz[3 * j + 1];
      sz[e] = support_xyz[3 * j + 2];
      sb[e] = batch_index[j];
    }
    if (base + TILE < NPTS)  // hint next tile into cache (global_prefetch_b8)
      __builtin_prefetch(&support_xyz[3 * (base + TILE) + 3 * lane], 0, 1);
    __syncthreads();
    for (int e = lane; e < TILE; e += 32) {
      if (sb[e] != qb) continue;
      float x = sx[e], y = sy[e], z = sz[e];
      float ss = x * x + y * y + z * z;
      float dot = qx * x + qy * y + qz * z;
      float d = qq + ss - 2.f * dot;  // same formula as reference
      int j = base + e;
      if (d < bd[15] || (d == bd[15] && j < bi[15])) {
        float cd = d; int ci = j;
#pragma unroll
        for (int s = 0; s < 16; ++s) {
          bool ins = (cd < bd[s]) || (cd == bd[s] && ci < bi[s]);
          float td = bd[s]; int ti = bi[s];
          if (ins) { bd[s] = cd; bi[s] = ci; cd = td; ci = ti; }
        }
      }
    }
  }

#pragma unroll
  for (int s = 0; s < 16; ++s) {
    md[wave][lane * 16 + s] = bd[s];
    mi[wave][lane * 16 + s] = bi[s];
  }
  __syncthreads();

  int hp = 0;
  for (int r = 0; r < 16; ++r) {
    float hd = (hp < 16) ? md[wave][lane * 16 + hp] : FINF;
    int hi = (hp < 16) ? mi[wave][lane * 16 + hp] : 0x7fffffff;
    int hl = lane;
#pragma unroll
    for (int off = 16; off >= 1; off >>= 1) {
      float od = __shfl_xor(hd, off, 32);
      int oi = __shfl_xor(hi, off, 32);
      int ol = __shfl_xor(hl, off, 32);
      if (od < hd || (od == hd && oi < hi)) { hd = od; hi = oi; hl = ol; }
    }
    if (lane == hl) hp++;
    if (lane == 0) col[q * 16 + r] = hi;
  }
}

// ---------------------------------------------------------------------------
// Kernel 2: per-query MLP + cosine affinities (all via v_wmma) + auction.
// One 128-thread (4-wave) workgroup per query.
// ---------------------------------------------------------------------------
__device__ inline void normalize_row128(const float* src, _Float16* dst) {
  float s = 0.f;
#pragma unroll 8
  for (int h = 0; h < 128; ++h) s = fmaf(src[h], src[h], s);
  float sc = 1.f / fmaxf(sqrtf(s), 1e-8f);
#pragma unroll 8
  for (int h = 0; h < 128; ++h) dst[h] = (_Float16)(src[h] * sc);
}

__global__ __launch_bounds__(128) void mat_kernel(
    const float* __restrict__ support_xyz, const int* __restrict__ filtered_index,
    const float* __restrict__ gcnfeat, const float* __restrict__ leaf,
    const float* __restrict__ b1, const float* __restrict__ b2,
    const float* __restrict__ bl, const int* __restrict__ col,
    const _Float16* __restrict__ W1h, const _Float16* __restrict__ W2h,
    const _Float16* __restrict__ Wlh, float* __restrict__ out) {
  __shared__ _Float16 sA1[16 * 32];      // padded edge features (A, layer1)
  __shared__ _Float16 sH1[16 * 64];      // relu hidden (A, layer2)
  __shared__ float sEW[16 * 128];        // edge_w f32
  __shared__ float sCW[16 * 128];        // cls_w  f32
  __shared__ float sFH[16 * 128];        // feat_j[:, :H] f32
  __shared__ _Float16 sAcls[16 * 32];    // padded class features (A, cls)
  __shared__ _Float16 sEWn[16 * 128];    // normalized (B^T sources)
  __shared__ _Float16 sCWn[16 * 128];
  __shared__ _Float16 sFHn[16 * 128];
  __shared__ _Float16 sLn[3][16 * 128];  // normalized leaf slices (A), rows 8..15 = 0
  __shared__ float sM[3][16 * 16];       // m1,m2,m3
  __shared__ float sMat[8 * 16];
  __shared__ int scol[16];

  int q = blockIdx.x;
  int t = threadIdx.x;
  int wave = t >> 5, lane = t & 31;

  // zero padded regions
  for (int e = t; e < 16 * 32; e += 128) { sA1[e] = (_Float16)0.f; sAcls[e] = (_Float16)0.f; }
  for (int e = t; e < 3 * 16 * 128; e += 128) (&sLn[0][0])[e] = (_Float16)0.f;
  if (t < 16) scol[t] = col[q * 16 + t];
  __syncthreads();

  // ---- edge features ew = [xyz_i, xyz_j, diff, dist] ----
  if (t < 16) {
    int fi = filtered_index[q];
    float qx = support_xyz[3 * fi], qy = support_xyz[3 * fi + 1], qz = support_xyz[3 * fi + 2];
    int j = scol[t];
    float sxv = support_xyz[3 * j], syv = support_xyz[3 * j + 1], szv = support_xyz[3 * j + 2];
    float dx = qx - sxv, dy = qy - syv, dz = qz - szv;
    float dist = sqrtf(dx * dx + dy * dy + dz * dz);
    _Float16* r = sA1 + t * 32;
    r[0] = (_Float16)qx; r[1] = (_Float16)qy; r[2] = (_Float16)qz;
    r[3] = (_Float16)sxv; r[4] = (_Float16)syv; r[5] = (_Float16)szv;
    r[6] = (_Float16)dx; r[7] = (_Float16)dy; r[8] = (_Float16)dz;
    r[9] = (_Float16)dist;
  }
  // ---- gather feat_j: first 128 -> sFH (f32), last 20 -> sAcls (f16) ----
  for (int e = t; e < 16 * FD; e += 128) {
    int k = e / FD, c = e % FD;
    float v = gcnfeat[(long)scol[k] * FD + c];
    if (c < HD) sFH[k * 128 + c] = v;
    else sAcls[k * 32 + (c - HD)] = (_Float16)v;
  }
  __syncthreads();

  // ---- layer1: relu(EW(16x32) @ W1(32x64) + b1) -> sH1 f16; wave w -> tile w
  if (wave < 4) {
    v16h a = load_A16x32(sA1, 32);
    v16h b = load_B32x16_KN(W1h + wave * 16, 64);
    v8f c = {};
    c = wmma_f16(a, b, c);
    int n = lane & 15, roff = (lane < 16) ? 0 : 8;
    float bias = b1[wave * 16 + n];
#pragma unroll
    for (int v = 0; v < 8; ++v) {
      float h = c[v] + bias;
      sH1[(v + roff) * 64 + wave * 16 + n] = (_Float16)(h > 0.f ? h : 0.f);
    }
  }
  __syncthreads();

  // ---- layer2: H1(16x64) @ W2(64x128) + b2 -> sEW f32 (2 tiles/wave) ----
  for (int tt = wave; tt < 8; tt += 4) {
    v8f c = {};
#pragma unroll
    for (int kk = 0; kk < 64; kk += 32) {
      v16h a = load_A16x32(sH1 + kk, 64);
      v16h b = load_B32x16_KN(W2h + kk * 128 + tt * 16, 128);
      c = wmma_f16(a, b, c);
    }
    int n = lane & 15, roff = (lane < 16) ? 0 : 8;
    float bias = b2[tt * 16 + n];
#pragma unroll
    for (int v = 0; v < 8; ++v) sEW[(v + roff) * 128 + tt * 16 + n] = c[v] + bias;
  }
  // ---- cls: F(16x32) @ Wl(32x128) + bl -> sCW f32 (2 tiles/wave) ----
  for (int tt = wave; tt < 8; tt += 4) {
    v16h a = load_A16x32(sAcls, 32);
    v16h b = load_B32x16_KN(Wlh + tt * 16, 128);
    v8f c = {};
    c = wmma_f16(a, b, c);
    int n = lane & 15, roff = (lane < 16) ? 0 : 8;
    float bias = bl[tt * 16 + n];
#pragma unroll
    for (int v = 0; v < 8; ++v) sCW[(v + roff) * 128 + tt * 16 + n] = c[v] + bias;
  }
  __syncthreads();

  // ---- row normalization (cosine) ----
  if (t < 16) normalize_row128(sEW + t * 128, sEWn + t * 128);
  else if (t < 32) normalize_row128(sCW + (t - 16) * 128, sCWn + (t - 16) * 128);
  else if (t < 48) normalize_row128(sFH + (t - 32) * 128, sFHn + (t - 32) * 128);
  else if (t < 72) {
    int idx = t - 48, s = idx >> 3, l = idx & 7;
    const float* src = leaf + ((long)q * LROW + l) * (3 * HD) + s * HD;
    float nr = 0.f;
#pragma unroll 8
    for (int h = 0; h < 128; ++h) nr = fmaf(src[h], src[h], nr);
    float sc = 1.f / fmaxf(sqrtf(nr), 1e-8f);
#pragma unroll 8
    for (int h = 0; h < 128; ++h) sLn[s][l * 128 + h] = (_Float16)(src[h] * sc);
  }
  __syncthreads();

  // ---- affinities: m_s = Ln[s](16x128) @ Bn^T(128x16); wave s computes m_s
  if (wave < 3) {
    const _Float16* A = sLn[wave];
    const _Float16* B = (wave == 0) ? sEWn : (wave == 1) ? sCWn : sFHn;
    v8f c = {};
#pragma unroll
    for (int kk = 0; kk < 128; kk += 32) {
      v16h a = load_A16x32(A + kk, 128);
      v16h b = load_B32x16_NK(B + kk, 128);
      c = wmma_f16(a, b, c);
    }
    int n = lane & 15, roff = (lane < 16) ? 0 : 8;
#pragma unroll
    for (int v = 0; v < 8; ++v) sM[wave][(v + roff) * 16 + n] = c[v];
  }
  __syncthreads();

  {  // mat = (m1+m2)/4 + m3/2 for rows 0..7
    int l = t >> 4, k = t & 15;
    if (l < 8) sMat[l * 16 + k] = (sM[0][l * 16 + k] + sM[1][l * 16 + k]) * 0.25f
                                  + sM[2][l * 16 + k] * 0.5f;
  }
  __syncthreads();

  // ---- auction LAP on wave 0 (lanes = columns) ----
  if (wave == 0) {
    const float eps = 1.0f / (float)LROW;
    float cost = 0.f;
    int ass[8];
#pragma unroll
    for (int l = 0; l < 8; ++l) ass[l] = -1;

    for (int it = 0; it < 10000; ++it) {
      bool anyun = false;
#pragma unroll
      for (int l = 0; l < 8; ++l) anyun |= (ass[l] < 0);
      if (!anyun) break;

      float hbid = 0.f;
      int hbdr = -1;
      for (int l = 0; l < 8; ++l) {
        if (ass[l] >= 0) continue;
        float v = -FINF;
        if (lane < 16) v = sMat[l * 16 + lane] - cost;
        float v1 = v, v2 = -FINF;
        int i1 = lane;
#pragma unroll
        for (int off = 8; off >= 1; off >>= 1) {  // stays within lane<16 group
          float ov1 = __shfl_xor(v1, off, 32);
          int oi1 = __shfl_xor(i1, off, 32);
          float ov2 = __shfl_xor(v2, off, 32);
          if (ov1 > v1 || (ov1 == v1 && oi1 < i1)) {
            v2 = fmaxf(v1, ov2);
            v1 = ov1; i1 = oi1;
          } else {
            v2 = fmaxf(v2, ov1);
          }
        }
        float bid = v1 - v2 + eps;
        if (lane < 16 && lane == i1) {
          if (bid > hbid) { hbid = bid; hbdr = l; }
        }
      }
      unsigned long long mask = __ballot(lane < 16 && hbid > 0.f);
      if (lane < 16 && hbid > 0.f) cost += hbid;
#pragma unroll
      for (int l = 0; l < 8; ++l)
        if (ass[l] >= 0 && ((mask >> ass[l]) & 1ull)) ass[l] = -1;
      for (int k = 0; k < 16; ++k) {
        if ((mask >> k) & 1ull) {
          int b = __shfl(hbdr, k, 32);
          ass[b] = k;
        }
      }
    }

    if (lane == 0) {
      float s = 0.f;
#pragma unroll
      for (int l = 0; l < 8; ++l) {
        int a = ass[l] < 0 ? 0 : ass[l];
        s += sMat[l * 16 + a];
      }
      out[q] = s * (1.0f / (float)LROW);
    }
  }
}

// ---------------------------------------------------------------------------
extern "C" void kernel_launch(void* const* d_in, const int* in_sizes, int n_in,
                              void* d_out, int out_size, void* d_ws, size_t ws_size,
                              hipStream_t stream) {
  const float* support_xyz    = (const float*)d_in[0];
  const int*   batch_index    = (const int*)d_in[1];
  const int*   filtered_index = (const int*)d_in[2];
  const float* gcnfeat        = (const float*)d_in[3];
  const float* leaf           = (const float*)d_in[4];
  const float* W1             = (const float*)d_in[5];
  const float* b1             = (const float*)d_in[6];
  const float* W2             = (const float*)d_in[7];
  const float* b2             = (const float*)d_in[8];
  const float* Wl             = (const float*)d_in[9];
  const float* bl             = (const float*)d_in[10];
  float* out = (float*)d_out;

  // workspace layout
  int* col = (int*)d_ws;                                    // Q*K ints
  _Float16* W1h = (_Float16*)((char*)d_ws + (size_t)NQ * KNN * sizeof(int));
  _Float16* W2h = W1h + 32 * 64;
  _Float16* Wlh = W2h + 64 * 128;

  prep_kernel<<<16, 256, 0, stream>>>(W1, W2, Wl, W1h, W2h, Wlh);
  knn_kernel<<<NQ / 8, 256, 0, stream>>>(support_xyz, batch_index, filtered_index, col);
  mat_kernel<<<NQ, 128, 0, stream>>>(support_xyz, filtered_index, gcnfeat, leaf,
                                     b1, b2, bl, col, W1h, W2h, Wlh, out);
}
